// MoDRouter_11192684773445
// MI455X (gfx1250) — matrix operand order repaired and checked by reading, not compiled, not run on hardware
//
#include <hip/hip_runtime.h>
#include <math.h>

#define BATCH 4
#define SEQL  8192
#define DIM   1024
#define KSEL  6144   // int(8192 * 0.75)

#define BM 64
#define BN 128
#define BK 32

typedef __attribute__((ext_vector_type(16))) __bf16 v16bf;
typedef __attribute__((ext_vector_type(8)))  __bf16 v8bf;
typedef __attribute__((ext_vector_type(8)))  float  v8f;

union BF16x16 { v16bf v; v8bf h[2]; };

__device__ __forceinline__ unsigned short f2bf(float f) {
    unsigned u = __float_as_uint(f);
    unsigned r = u + 0x7FFFu + ((u >> 16) & 1u);   // round-to-nearest-even
    return (unsigned short)(r >> 16);
}

// monotone float -> uint mapping (order preserving)
__device__ __forceinline__ unsigned orderKey(float f) {
    unsigned u = __float_as_uint(f);
    return (u & 0x80000000u) ? ~u : (u | 0x80000000u);
}

__device__ __forceinline__ float gelu_tanh(float v) {
    // jax.nn.gelu default: approximate=True (tanh form)
    return 0.5f * v * (1.0f + tanhf(0.7978845608028654f * (v + 0.044715f * v * v * v)));
}

// CDNA5 async copy: global -> LDS, ASYNCcnt-tracked.
// LDS target comes from VDST, global source from VADDR; both add the immediate offset
__device__ __forceinline__ void async_b128(unsigned ldsOff, const void* gaddr) {
    asm volatile("global_load_async_to_lds_b128 %0, %1, off"
                 :: "v"(ldsOff), "v"(gaddr) : "memory");
}
__device__ __forceinline__ void async_b128x2(unsigned ldsOff, const void* gaddr) {
    asm volatile("global_load_async_to_lds_b128 %0, %1, off\n\t"
                 "global_load_async_to_lds_b128 %0, %1, off offset:16"
                 :: "v"(ldsOff), "v"(gaddr) : "memory");
}
__device__ __forceinline__ void wait_async0() {
    asm volatile("s_wait_asynccnt 0" ::: "memory");
}

// ---- 1) router scores + fp32->bf16 conversion of x (single pass over x) ---
__global__ __launch_bounds__(256) void router_convert(const float* __restrict__ x,
                                                      const float* __restrict__ wr,
                                                      float* __restrict__ scores,
                                                      unsigned short* __restrict__ xb) {
    int wave = threadIdx.x >> 5;
    int lane = threadIdx.x & 31;
    int t = blockIdx.x * 8 + wave;                 // token id over BATCH*SEQL
    const float* xp = x + (size_t)t * DIM;
    unsigned short* xo = xb + (size_t)t * DIM;
    float s = 0.f;
#pragma unroll
    for (int j = 0; j < 8; ++j) {
        int e = j * 128 + lane * 4;
        float4 xv = *(const float4*)(xp + e);
        float4 wv = *(const float4*)(wr + e);
        s += xv.x * wv.x + xv.y * wv.y + xv.z * wv.z + xv.w * wv.w;
        uint2 p;
        p.x = (unsigned)f2bf(xv.x) | ((unsigned)f2bf(xv.y) << 16);
        p.y = (unsigned)f2bf(xv.z) | ((unsigned)f2bf(xv.w) << 16);
        *(uint2*)(xo + e) = p;
    }
#pragma unroll
    for (int m = 16; m >= 1; m >>= 1) s += __shfl_xor(s, m, 32);
    if (lane == 0) scores[t] = s;
}

// ------- 2) exact k-th largest per batch row via 32-step bit descent -------
__global__ __launch_bounds__(256) void select_threshold(const float* __restrict__ scores,
                                                        unsigned* __restrict__ thr,
                                                        unsigned* __restrict__ cnts) {
    __shared__ unsigned s_cnt;
    int b = blockIdx.x, t = threadIdx.x;
    unsigned key[32];
#pragma unroll
    for (int j = 0; j < 32; ++j)
        key[j] = orderKey(scores[b * SEQL + j * 256 + t]);

    unsigned T = 0;
    for (int bit = 31; bit >= 0; --bit) {
        unsigned cand = T | (1u << bit);
        if (t == 0) s_cnt = 0;
        __syncthreads();
        unsigned c = 0;
#pragma unroll
        for (int j = 0; j < 32; ++j) c += (key[j] >= cand) ? 1u : 0u;
#pragma unroll
        for (int m = 16; m >= 1; m >>= 1) c += __shfl_xor(c, m, 32);
        if ((t & 31) == 0) atomicAdd(&s_cnt, c);
        __syncthreads();
        if (s_cnt >= KSEL) T = cand;               // max T with count(>=T) >= k
        __syncthreads();
    }
    if (t == 0) { thr[b] = T; cnts[b] = 0; }
}

// ---------------- 3) W -> bf16, transposed to [N][K] -----------------------
__global__ __launch_bounds__(256) void convert_wT(const float* __restrict__ w,
                                                  unsigned short* __restrict__ wbT) {
    int tid = blockIdx.x * 256 + threadIdx.x;      // over DIM*DIM/4
    int n  = tid / (DIM / 4);
    int k4 = (tid % (DIM / 4)) * 4;
    unsigned p0 = (unsigned)f2bf(w[(size_t)(k4 + 0) * DIM + n]) |
                  ((unsigned)f2bf(w[(size_t)(k4 + 1) * DIM + n]) << 16);
    unsigned p1 = (unsigned)f2bf(w[(size_t)(k4 + 2) * DIM + n]) |
                  ((unsigned)f2bf(w[(size_t)(k4 + 3) * DIM + n]) << 16);
    uint2 v; v.x = p0; v.y = p1;
    *(uint2*)(wbT + (size_t)n * DIM + k4) = v;
}

// ---- 4) compact selected token indices; pass-through copy for the rest ----
__global__ __launch_bounds__(256) void compact_copy(const float* __restrict__ x,
                                                    const float* __restrict__ scores,
                                                    const unsigned* __restrict__ thr,
                                                    unsigned* __restrict__ cnts,
                                                    int* __restrict__ idx,
                                                    float* __restrict__ out) {
    int wave = threadIdx.x >> 5, lane = threadIdx.x & 31;
    int t = blockIdx.x * 8 + wave;
    int b = t / SEQL, l = t % SEQL;
    int sel = 0;
    if (lane == 0) {
        sel = (orderKey(scores[t]) >= thr[b]) ? 1 : 0;
        if (sel) {
            unsigned slot = atomicAdd(&cnts[b], 1u);
            if (slot < KSEL) idx[b * KSEL + slot] = l;
        }
    }
    sel = __shfl(sel, 0, 32);
    if (!sel) {
        const float4* src = (const float4*)(x + (size_t)t * DIM);
        float4* dst = (float4*)(out + (size_t)t * DIM);
#pragma unroll
        for (int j = 0; j < 8; ++j) dst[j * 32 + lane] = src[j * 32 + lane];
    }
}

// -- 5) gathered GEMM + GELU: bf16 WMMA, double-buffered async-LDS staging --
__global__ __launch_bounds__(256) void mod_gemm(const unsigned short* __restrict__ xb,
                                                const unsigned short* __restrict__ wbT,
                                                const int* __restrict__ idx,
                                                float* __restrict__ out) {
    __shared__ int rows_s[BM];
    __shared__ __align__(64) __bf16 aT[2][BM][BK];   // 2 x 4 KB
    __shared__ __align__(64) __bf16 bT[2][BN][BK];   // 2 x 8 KB

    int b  = blockIdx.z;
    int m0 = blockIdx.y * BM;
    int n0 = blockIdx.x * BN;
    int tid = threadIdx.x;
    int wave = tid >> 5, lane = tid & 31;
    int waveM = wave & 3, waveN = wave >> 2;         // 4 x 2 wave layout
    int laneHalf = lane >> 4;                        // 0: lanes 0-15, 1: lanes 16-31
    int lm = lane & 15;

    if (tid < BM) rows_s[tid] = idx[b * KSEL + m0 + tid];
    __syncthreads();

    // staging coordinates (per thread: 16B of A, 32B of B)
    int ar = tid >> 2;                               // A row 0..63
    int ac = (tid & 3) * 8;                          // A k-chunk {0,8,16,24}
    const unsigned short* aBase = xb + ((size_t)b * SEQL + rows_s[ar]) * DIM + ac;
    int br = tid >> 1;                               // B col 0..127
    int bc = (tid & 1) * 16;                         // B k-chunk {0,16}
    const unsigned short* bBase = wbT + (size_t)(n0 + br) * DIM + bc;

    // LDS byte offsets for async VDST (flat shared addr truncates to LDS offset)
    unsigned aDst[2] = { (unsigned)(uintptr_t)&aT[0][ar][ac],
                         (unsigned)(uintptr_t)&aT[1][ar][ac] };
    unsigned bDst[2] = { (unsigned)(uintptr_t)&bT[0][br][bc],
                         (unsigned)(uintptr_t)&bT[1][br][bc] };

    // prologue: stage tile 0 into buffer 0
    async_b128 (aDst[0], aBase);
    async_b128x2(bDst[0], bBase);

    v8f acc[4] = {};
    int arow = waveM * 16 + lm;
    int alo  = laneHalf * 16;
    int bko  = laneHalf * 16;

    for (int k0 = 0, it = 0; k0 < DIM; k0 += BK, ++it) {
        int cur = it & 1;
        wait_async0();          // my loads for buffer `cur` are in LDS
        __syncthreads();        // everyone's loads done; prev compute reads done
        if (k0 + BK < DIM) {    // stage next tile into the other buffer
            async_b128 (aDst[cur ^ 1], aBase + k0 + BK);
            async_b128x2(bDst[cur ^ 1], bBase + k0 + BK);
        }

        // A operand: lanes 0-15 K 0..15, 16-31 K 16..31 (matches B)
        BF16x16 A;
        A.h[0] = *(const v8bf*)&aT[cur][arow][alo];
        A.h[1] = *(const v8bf*)&aT[cur][arow][alo + 8];

#pragma unroll
        for (int nt = 0; nt < 4; ++nt) {
            // B operand: lane n holds column n; lanes 0-15 K 0..15, 16-31 K 16..31
            int bcol = waveN * 64 + nt * 16 + lm;
            v16bf Bv = *(const v16bf*)&bT[cur][bcol][bko];   // 32 contiguous bytes
            acc[nt] = __builtin_amdgcn_wmma_f32_16x16x32_bf16(
                false, A.v, false, Bv, (short)0, acc[nt], false, false);
        }
    }

    // epilogue: GELU + scatter store through gathered row indices
#pragma unroll
    for (int nt = 0; nt < 4; ++nt) {
        int col = n0 + waveN * 64 + nt * 16 + lm;
#pragma unroll
        for (int r = 0; r < 8; ++r) {
            int rowLocal = waveM * 16 + r + laneHalf * 8;   // C layout: vgpr r -> M=r / r+8
            int l = rows_s[rowLocal];
            out[((size_t)b * SEQL + l) * DIM + col] = gelu_tanh(acc[nt][r]);
        }
    }
}

extern "C" void kernel_launch(void* const* d_in, const int* in_sizes, int n_in,
                              void* d_out, int out_size, void* d_ws, size_t ws_size,
                              hipStream_t stream) {
    (void)in_sizes; (void)n_in; (void)out_size; (void)ws_size;
    const float* x  = (const float*)d_in[0];
    const float* wr = (const float*)d_in[1];
    const float* wb = (const float*)d_in[2];
    float* out = (float*)d_out;

    char* ws = (char*)d_ws;
    float*          scores = (float*)(ws + 0);                 // 128 KB
    unsigned*       thr    = (unsigned*)(ws + 131072);
    unsigned*       cnts   = (unsigned*)(ws + 131328);
    int*            idx    = (int*)(ws + 131584);              // 96 KB
    unsigned short* wbT    = (unsigned short*)(ws + 262144);   // 2 MB
    unsigned short* xbuf   = (unsigned short*)(ws + 2359296);  // 64 MB bf16 x

    router_convert<<<(BATCH * SEQL) / 8, 256, 0, stream>>>(x, wr, scores, xbuf);
    select_threshold<<<BATCH, 256, 0, stream>>>(scores, thr, cnts);
    convert_wT<<<(DIM * DIM / 4) / 256, 256, 0, stream>>>(wb, wbT);
    compact_copy<<<(BATCH * SEQL) / 8, 256, 0, stream>>>(x, scores, thr, cnts, idx, out);
    mod_gemm<<<dim3(DIM / BN, KSEL / BM, BATCH), 256, 0, stream>>>(xbuf, wbT, idx, out);
}
